// GlobalAttention_68959994904654
// MI455X (gfx1250) — matrix operand order repaired and checked by reading, not compiled, not run on hardware
//
#include <hip/hip_runtime.h>

#define DIM   1024
#define SEQ   2048
#define BATCH 32
#define MBLK  64
#define A_STR 1032   // bf16 elements per padded LDS row (2064 B = 129*16B, bank-conflict free)

typedef __attribute__((ext_vector_type(16))) __bf16 v16bf;
typedef __attribute__((ext_vector_type(8)))  float  v8f;
typedef __attribute__((ext_vector_type(4)))  int    v4i;

union Frag16 {
  uint4 u[2];
  v16bf v;
};

__device__ __forceinline__ unsigned short f2bf(float f) {
  unsigned int u = __float_as_uint(f);
  u += 0x7fffu + ((u >> 16) & 1u);   // round-to-nearest-even
  return (unsigned short)(u >> 16);
}

// ---- CDNA5 async global->LDS path (ASYNCcnt), guarded so we never regress ----
#if defined(__HIP_DEVICE_COMPILE__) && defined(__gfx1250__) && \
    __has_builtin(__builtin_amdgcn_global_load_async_to_lds_b128)
#define USE_ASYNC_LDS 1
#else
#define USE_ASYNC_LDS 0
#endif

#if USE_ASYNC_LDS
__device__ __forceinline__ void async_cp16(const void* g, void* l) {
  // per-lane 16B copy: global_load_async_to_lds_b128 (bypasses VGPRs, ASYNCcnt)
  __builtin_amdgcn_global_load_async_to_lds_b128((v4i*)g, (v4i*)l, 0, 0);
}
#endif

__device__ __forceinline__ void wait_async0() {
#if defined(__HIP_DEVICE_COMPILE__) && defined(__gfx1250__) && \
    __has_builtin(__builtin_amdgcn_s_wait_asynccnt)
  __builtin_amdgcn_s_wait_asynccnt(0);
#elif defined(__HIP_DEVICE_COMPILE__) && defined(__gfx1250__)
  asm volatile("s_wait_asynccnt 0x0" ::: "memory");
#endif
}

// ---------------------------------------------------------------------------
// 1. Convert Wa_s = Wa[:, D:2D] (fp32, row stride 2D) -> bf16, row-major KxD.
// ---------------------------------------------------------------------------
__global__ void __launch_bounds__(256)
convert_was_kernel(const float* __restrict__ Wa, unsigned short* __restrict__ WaS) {
  int i  = blockIdx.x * blockDim.x + threadIdx.x;  // float4 index over DxD
  int fl = i << 2;
  int row = fl >> 10;
  int col = fl & 1023;
  float4 f = *(const float4*)(Wa + (size_t)row * (2 * DIM) + DIM + col);
  uint2 p;
  p.x = (unsigned)f2bf(f.x) | ((unsigned)f2bf(f.y) << 16);
  p.y = (unsigned)f2bf(f.z) | ((unsigned)f2bf(f.w) << 16);
  *(uint2*)(WaS + (size_t)row * DIM + col) = p;
}

// ---------------------------------------------------------------------------
// 2. ht_proj[b,n] = sum_d x[b,d] * Wa[n,d]   (tiny: 67 MFLOP)
// ---------------------------------------------------------------------------
__global__ void __launch_bounds__(128)
ht_proj_kernel(const float* __restrict__ x, const float* __restrict__ Wa,
               float* __restrict__ ht) {
  __shared__ float xs[BATCH][129];
  const int tid = threadIdx.x;
  const int n   = blockIdx.x * 128 + tid;
  const float* warow = Wa + (size_t)n * (2 * DIM);
  float acc[BATCH];
#pragma unroll
  for (int bb = 0; bb < BATCH; ++bb) acc[bb] = 0.f;

  for (int d0 = 0; d0 < DIM; d0 += 128) {
    __syncthreads();
    for (int i = tid; i < BATCH * 128; i += 128) {
      int bb = i >> 7, cc = i & 127;
      xs[bb][cc] = x[(size_t)bb * DIM + d0 + cc];
    }
    __syncthreads();
    for (int c = 0; c < 128; ++c) {
      float wv = warow[d0 + c];
#pragma unroll
      for (int bb = 0; bb < BATCH; ++bb) acc[bb] += xs[bb][c] * wv;
    }
  }
#pragma unroll
  for (int bb = 0; bb < BATCH; ++bb) ht[(size_t)bb * DIM + n] = acc[bb];
}

// ---------------------------------------------------------------------------
// 3. Fused scores GEMM: scores[b,l] = sum_n Va[n]*tanh(ht[b,n] + ctx[b,l,:]·WaS[n,:])
//    Block: one b, 64 l-rows, 4 waves (one 16-row M-tile each).
//    A (context tile) fp32->bf16 staged in LDS, preloaded to VGPR fragments.
//    B (Wa_s tile) double-buffered in LDS via async global->LDS copies.
//    Inner loop: 32 chained v_wmma_f32_16x16x32_bf16 over K=1024.
// ---------------------------------------------------------------------------
__device__ __forceinline__ void stage_b_tile(const unsigned short* WaS, int nt,
                                             unsigned short* dst, int tid) {
  const uint4* wsrc = (const uint4*)(WaS + (size_t)nt * 16 * DIM);
  for (int i = tid; i < 16 * DIM / 8; i += 128) {   // 2048 x 16B
    int row = i >> 7;            // 128 uint4 per row
    int c8  = i & 127;
#if USE_ASYNC_LDS
    async_cp16(wsrc + i, dst + row * A_STR + (c8 << 3));
#else
    uint4 vv = wsrc[i];
    *(uint4*)(dst + row * A_STR + (c8 << 3)) = vv;
#endif
  }
}

__global__ void __launch_bounds__(128)
attn_scores_kernel(const float* __restrict__ ctx,
                   const unsigned short* __restrict__ WaS,
                   const float* __restrict__ ht,
                   const float* __restrict__ Va,
                   float* __restrict__ scores) {
  extern __shared__ __align__(16) unsigned short lds[];
  unsigned short* ldsA = lds;                              // 64 x A_STR bf16
  unsigned short* ldsB0 = lds + MBLK * A_STR;              // 16 x A_STR bf16
  unsigned short* ldsB1 = ldsB0 + 16 * A_STR;              // 16 x A_STR bf16

  const int tid  = threadIdx.x;
  const int wave = tid >> 5;
  const int lane = tid & 31;
  const int half = lane >> 4;     // which 16-lane half
  const int l16  = lane & 15;
  const int b    = blockIdx.y;
  const int l0   = blockIdx.x * MBLK;

  // ---- kick off async staging of the first B tile immediately
  stage_b_tile(WaS, 0, ldsB0, tid);

  // ---- stage A: context[b, l0:l0+64, :] fp32 -> bf16 LDS (rows contiguous)
  const float4* src = (const float4*)(ctx + ((size_t)b * SEQ + l0) * DIM);
  for (int i = tid; i < MBLK * DIM / 4; i += 128) {
    float4 f = src[i];
    int row = i >> 8;             // DIM/4 = 256 float4 per row
    int c4  = i & 255;
    uint2 p;
    p.x = (unsigned)f2bf(f.x) | ((unsigned)f2bf(f.y) << 16);
    p.y = (unsigned)f2bf(f.z) | ((unsigned)f2bf(f.w) << 16);
    *(uint2*)(ldsA + row * A_STR + (c4 << 2)) = p;
  }
  wait_async0();
  __syncthreads();

  // ---- preload this wave's A fragments (ISA 16-bit A 16x32 layout):
  // lanes 0-15: K = 32k+{0..7} | 32k+{16..23}; lanes 16-31: +8.
  Frag16 afrag[32];
  const unsigned short* arow = ldsA + (wave * 16 + l16) * A_STR;
#pragma unroll
  for (int kc = 0; kc < 32; ++kc) {
    afrag[kc].u[0] = *(const uint4*)(arow + kc * 32 +      half * 8);
    afrag[kc].u[1] = *(const uint4*)(arow + kc * 32 + 16 + half * 8);
  }

  float acc[8];
#pragma unroll
  for (int r = 0; r < 8; ++r) acc[r] = 0.f;

  for (int nt = 0; nt < DIM / 16; ++nt) {
    unsigned short* cur = (nt & 1) ? ldsB1 : ldsB0;
    // prefetch next B tile into the alternate buffer while we compute
    if (nt + 1 < DIM / 16)
      stage_b_tile(WaS, nt + 1, (nt & 1) ? ldsB0 : ldsB1, tid);

    v8f c = {0.f, 0.f, 0.f, 0.f, 0.f, 0.f, 0.f, 0.f};
#pragma unroll
    for (int kc = 0; kc < 32; ++kc) {
      // B 32x16 bf16 layout: lane n=l16, K = 32k+{0..7} | 32k+{16..23}, +8 for upper half
      Frag16 bf;
      const unsigned short* brow = cur + l16 * A_STR + kc * 32 + half * 8;
      bf.u[0] = *(const uint4*)(brow);
      bf.u[1] = *(const uint4*)(brow + 16);
      c = __builtin_amdgcn_wmma_f32_16x16x32_bf16(false, afrag[kc].v,
                                                  false, bf.v,
                                                  (short)0, c, false, false);
    }

    // ---- fused epilogue: C layout -> m = 8*half + r, n = nt*16 + l16
    const int   n   = nt * 16 + l16;
    const float htv = ht[(size_t)b * DIM + n];
    const float vav = Va[n];
#pragma unroll
    for (int r = 0; r < 8; ++r)
      acc[r] += vav * tanhf(c[r] + htv);

    wait_async0();    // our async writes into the alternate buffer are done
    __syncthreads();  // everyone's are visible before next iteration reads it
  }

  // ---- reduce over the 16 n-columns held per half-wave
#pragma unroll
  for (int r = 0; r < 8; ++r) {
    float v = acc[r];
    v += __shfl_xor(v, 1, 32);
    v += __shfl_xor(v, 2, 32);
    v += __shfl_xor(v, 4, 32);
    v += __shfl_xor(v, 8, 32);
    if (l16 == 0) {
      int m = wave * 16 + half * 8 + r;
      scores[(size_t)b * SEQ + l0 + m] = v;
    }
  }
}

// ---------------------------------------------------------------------------
// 4. Per-batch softmax over L=2048 -> attn output (keeps exp() in ws).
// ---------------------------------------------------------------------------
__global__ void __launch_bounds__(256)
softmax_kernel(float* __restrict__ scores, float* __restrict__ attn) {
  __shared__ float red[256];
  const int b = blockIdx.x, tid = threadIdx.x;
  float* srow = scores + (size_t)b * SEQ;
  float m = -3.4e38f;
  for (int l = tid; l < SEQ; l += 256) m = fmaxf(m, srow[l]);
  red[tid] = m; __syncthreads();
  for (int st = 128; st > 0; st >>= 1) {
    if (tid < st) red[tid] = fmaxf(red[tid], red[tid + st]);
    __syncthreads();
  }
  const float mx = red[0];
  __syncthreads();
  float s = 0.f;
  for (int l = tid; l < SEQ; l += 256) {
    float e = __expf(srow[l] - mx);
    srow[l] = e;
    s += e;
  }
  red[tid] = s; __syncthreads();
  for (int st = 128; st > 0; st >>= 1) {
    if (tid < st) red[tid] += red[tid + st];
    __syncthreads();
  }
  const float inv = 1.f / red[0];
  for (int l = tid; l < SEQ; l += 256)
    attn[(size_t)b * SEQ + l] = srow[l] * inv;
}

// ---------------------------------------------------------------------------
// 5. weighted[b,d] = sum_l attn[b,l]*ctx[b,l,d]  (memory-bound, l-split + atomics)
// ---------------------------------------------------------------------------
__global__ void __launch_bounds__(256)
zero_kernel(float* __restrict__ p, int n) {
  int i = blockIdx.x * blockDim.x + threadIdx.x;
  if (i < n) p[i] = 0.f;
}

__global__ void __launch_bounds__(256)
weighted_kernel(const float* __restrict__ ctx, const float* __restrict__ attn,
                float* __restrict__ out) {
  const int b   = blockIdx.x;
  const int lc  = blockIdx.y;   // 8 chunks of 256 l's
  const int tid = threadIdx.x;
  const float4* ctx4 = (const float4*)(ctx + ((size_t)b * SEQ + lc * 256) * DIM);
  const float*  arow = attn + (size_t)b * SEQ + lc * 256;
  float4 acc = {0.f, 0.f, 0.f, 0.f};
  for (int l = 0; l < 256; ++l) {
    float  a = arow[l];
    float4 c = ctx4[(size_t)l * (DIM / 4) + tid];
    acc.x += a * c.x; acc.y += a * c.y; acc.z += a * c.z; acc.w += a * c.w;
  }
  float* o = out + (size_t)b * DIM + tid * 4;
  atomicAdd(o + 0, acc.x);
  atomicAdd(o + 1, acc.y);
  atomicAdd(o + 2, acc.z);
  atomicAdd(o + 3, acc.w);
}

// ---------------------------------------------------------------------------
extern "C" void kernel_launch(void* const* d_in, const int* in_sizes, int n_in,
                              void* d_out, int out_size, void* d_ws, size_t ws_size,
                              hipStream_t stream) {
  const float* x   = (const float*)d_in[0];
  const float* ctx = (const float*)d_in[1];
  const float* Wa  = (const float*)d_in[2];
  const float* Va  = (const float*)d_in[3];

  float* out_weighted = (float*)d_out;                 // 32*1024
  float* out_attn     = out_weighted + BATCH * DIM;    // 32*2048

  // workspace: WaS bf16 (2MB) | ht (128KB) | scores (256KB)
  unsigned short* WaS = (unsigned short*)d_ws;
  float* ht     = (float*)(WaS + (size_t)DIM * DIM);
  float* scores = ht + (size_t)BATCH * DIM;

  convert_was_kernel<<<(DIM * DIM / 4) / 256, 256, 0, stream>>>(Wa, WaS);

  ht_proj_kernel<<<DIM / 128, 128, 0, stream>>>(x, Wa, ht);

  // A tile (64 rows) + double-buffered B tiles (2 x 16 rows): 198,144 B of the
  // 320KB/WGP CDNA5 LDS.
  const int lds_bytes = (MBLK + 32) * A_STR * 2;
  (void)hipFuncSetAttribute((const void*)attn_scores_kernel,
                            hipFuncAttributeMaxDynamicSharedMemorySize, lds_bytes);
  attn_scores_kernel<<<dim3(SEQ / MBLK, BATCH), 128, lds_bytes, stream>>>(
      ctx, WaS, ht, Va, scores);

  softmax_kernel<<<BATCH, 256, 0, stream>>>(scores, out_attn);

  zero_kernel<<<(BATCH * DIM + 255) / 256, 256, 0, stream>>>(out_weighted, BATCH * DIM);
  weighted_kernel<<<dim3(BATCH, SEQ / 256), 256, 0, stream>>>(ctx, out_attn, out_weighted);
}
